// DirectPathAttenuationGNN_60687887892634
// MI455X (gfx1250) — matrix operand order repaired and checked by reading, not compiled, not run on hardware
//
#include <hip/hip_runtime.h>
#include <hip/hip_bf16.h>
#include <math.h>

typedef _Float16 h16;
typedef __attribute__((ext_vector_type(16))) _Float16 v16h;
typedef __attribute__((ext_vector_type(8)))  _Float16 v8h;
typedef __attribute__((ext_vector_type(8)))  float    v8f;

#define SENS   9
#define EPG    72          // edges per graph
#define GPB    2           // graphs per workgroup
#define EB     (EPG*GPB)   // 144 edges per workgroup (9 M-tiles)
#define NB     (SENS*GPB)  // 18 nodes per workgroup
#define NPAD   32          // node rows padded to 2 M-tiles
#define HDIM   128
#define LAYERS 4
#define LDH    136         // padded LDS row stride (halfs)
#define EPSV   1e-8f

// ---- workspace layout (in halfs): fragment-packed f16 weights ----
#define OFF_ENCW2 0                         // [128,128]       -> 16384
#define OFF_EW1   16384                     // 4x[384,128]     -> 196608
#define OFF_EW2   (OFF_EW1 + 196608)        // 4x[128,128]     -> 65536
#define OFF_NW1   (OFF_EW2 + 65536)         // 4x[256,128]     -> 131072
#define OFF_NW2   (OFF_NW1 + 131072)        // 4x[128,128]     -> 65536
#define OFF_DW1   (OFF_NW2 + 65536)         // [128,64]        -> 8192

// ---- LDS layout (bytes) ----
#define L_HE    0                            // h16 [144][LDH]  39168
#define L_TMP   39168                        // h16 [144][LDH]  39168
#define L_HN    78336                        // h16 [32][LDH]    8704
#define L_AGG   87040                        // h16 [32][LDH]    8704
#define L_PHYS  95744                        // f32 [144][8]     4608
#define L_SRC   100352                       // int [72]          288
#define L_DST   100640                       // int [72]          288
#define L_DEC   100928                       // f32 [144]         576
#define SMEM_BYTES 101504

// Pack W[K][N] row-major f32 (L layers) into B-fragment order for
// v_wmma_f32_16x16x32_f16: frag(kt,nt) = 512 halfs, half index = lane*16+i,
// lane<16 -> (k = kt*32+(i<8 ? i : 16+i-8),    n = nt*16+lane)
// lane>=16 -> (k = kt*32+8+(i<8 ? i : 16+i-8), n = nt*16+lane-16)
__global__ void pack_w_kernel(const float* __restrict__ src, h16* __restrict__ dst,
                              int K, int Nn, int layers) {
    int nnt = Nn >> 4;
    int perLayer = (K >> 5) * nnt * 512;
    int total = layers * perLayer;
    for (int idx = blockIdx.x * blockDim.x + threadIdx.x; idx < total;
         idx += gridDim.x * blockDim.x) {
        int l = idx / perLayer, r = idx % perLayer;
        int frag = r >> 9, hh = r & 511;
        int kt = frag / nnt, nt = frag % nnt;
        int lane = hh >> 4, i = hh & 15;
        int k = kt * 32 + ((i & 8) ? 16 : 0) + ((lane & 16) ? 8 : 0) + (i & 7);
        int n = nt * 16 + (lane & 15);
        dst[idx] = (h16)src[(size_t)l * K * Nn + k * Nn + n];
    }
}

// A fragment (16x32 f16) from an LDS row-major matrix:
// lane<16: halfs = row[kb+0..7], row[kb+16..23]; lane>=16: +8 offset
static __device__ __forceinline__ v16h loadA(const h16* rowp, int kb, int lane) {
    int o = kb + ((lane & 16) ? 8 : 0);
    union { v16h v; v8h h[2]; } u;
    u.h[0] = *(const v8h*)(rowp + o);
    u.h[1] = *(const v8h*)(rowp + o + 16);
    return u.v;
}
static __device__ __forceinline__ v16h loadB(const h16* frag, int lane) {
    return *(const v16h*)(frag + lane * 16);
}
static __device__ __forceinline__ v8f wmma16(v16h a, v16h b, v8f c) {
    return __builtin_amdgcn_wmma_f32_16x16x32_f16(false, a, false, b,
                                                  (short)0, c, false, false);
}

__global__ __launch_bounds__(256) void gnn_fused_kernel(
    const float* __restrict__ xn,    const float* __restrict__ dmg,
    const float* __restrict__ encNw, const float* __restrict__ encNb,
    const float* __restrict__ encEw1,const float* __restrict__ encEb1,
    const float* __restrict__ encEb2,
    const float* __restrict__ eb1,   const float* __restrict__ eb2,
    const float* __restrict__ nb1,   const float* __restrict__ nb2,
    const float* __restrict__ db1,   const float* __restrict__ dw2,
    const float* __restrict__ db2,
    const h16* __restrict__ encW2p,  const h16* __restrict__ eW1p,
    const h16* __restrict__ eW2p,    const h16* __restrict__ nW1p,
    const h16* __restrict__ nW2p,    const h16* __restrict__ dW1p,
    float* __restrict__ out) {
    extern __shared__ char smem[];
    h16*   he   = (h16*)(smem + L_HE);
    h16*   tmp  = (h16*)(smem + L_TMP);
    h16*   hn   = (h16*)(smem + L_HN);
    h16*   agg  = (h16*)(smem + L_AGG);
    float* phys = (float*)(smem + L_PHYS);
    int*   srcN = (int*)(smem + L_SRC);
    int*   dstN = (int*)(smem + L_DST);
    float* decL = (float*)(smem + L_DEC);

    const int tid  = threadIdx.x;
    const int lane = tid & 31;
    const int wv   = tid >> 5;             // 8 wave32 == 8 N-tiles
    const int gb   = blockIdx.x * GPB;     // first graph of this block
    const int lnlo = lane & 15;
    const int rsh  = (lane & 16) ? 8 : 0;  // C-fragment row offset per lane half

    // ---- within-graph edge topology table (matches _build_edge_index) ----
    if (tid == 0) {
        int t = 0;
        for (int i = 0; i < SENS; ++i)
            for (int j = i + 1; j < SENS; ++j) {
                srcN[t] = i; dstN[t] = j; ++t;
                srcN[t] = j; dstN[t] = i; ++t;
            }
    }
    __syncthreads();

    // ---- physics features per edge (VALU, tiny) ----
    for (int e = tid; e < EB; e += 256) {
        int g = e / EPG, t = e % EPG;
        int base = (gb + g) * SENS;
        const float* sp = xn + (base + srcN[t]) * 2;
        const float* dp = xn + (base + dstN[t]) * 2;
        float sx = sp[0], sy = sp[1], dx = dp[0], dy = dp[1];
        float mx = dmg[(gb + g) * 2 + 0], my = dmg[(gb + g) * 2 + 1];
        float vx = sx - dx, vy = sy - dy;
        float elen = sqrtf(vx * vx + vy * vy + EPSV);
        float ax = dx - sx, ay = dy - sy;
        float l2 = fmaxf(ax * ax + ay * ay, EPSV);
        float tt = ((mx - sx) * ax + (my - sy) * ay) / l2;
        tt = fminf(fmaxf(tt, 0.f), 1.f);
        float px = sx + tt * ax, py = sy + tt * ay;
        float dpath = sqrtf((mx - px) * (mx - px) + (my - py) * (my - py) + EPSV);
        float dtx = sqrtf((sx - mx) * (sx - mx) + (sy - my) * (sy - my) + EPSV);
        float drx = sqrtf((dx - mx) * (dx - mx) + (dy - my) * (dy - my) + EPSV);
        float* pr = phys + e * 8;
        pr[0] = vx; pr[1] = vy; pr[2] = elen; pr[3] = dpath; pr[4] = dtx; pr[5] = drx;
    }
    // ---- node encoder (K=2, VALU) + zero the padded rows of hn/agg ----
    for (int idx = tid; idx < NPAD * HDIM; idx += 256) {
        int r = idx / HDIM, c = idx % HDIM;
        float v = 0.f;
        if (r < NB) {
            int g = r / SENS, n = r % SENS;
            const float* xp = xn + ((gb + g) * SENS + n) * 2;
            v = xp[0] * encNw[c] + xp[1] * encNw[HDIM + c] + encNb[c];
        }
        hn[r * LDH + c] = (h16)v;
        agg[r * LDH + c] = (h16)0.f;
    }
    __syncthreads();

    // ---- edge encoder hidden: relu(phys @ W1[6,128] + b1) (K=6, VALU) ----
    for (int idx = tid; idx < EB * HDIM; idx += 256) {
        int e = idx / HDIM, c = idx % HDIM;
        const float* pr = phys + e * 8;
        float s = encEb1[c];
        #pragma unroll
        for (int k = 0; k < 6; ++k) s += pr[k] * encEw1[k * HDIM + c];
        tmp[e * LDH + c] = (h16)fmaxf(s, 0.f);
    }
    __syncthreads();

    // ---- h_e = hidden @ encW2[128,128] + b2 (WMMA; wave wv owns nt=wv) ----
    {
        const int nt = wv, col = nt * 16 + lnlo;
        v16h bf[4];
        #pragma unroll
        for (int ks = 0; ks < 4; ++ks) bf[ks] = loadB(encW2p + (ks * 8 + nt) * 512, lane);
        const float bv = encEb2[col];
        for (int mt = 0; mt < 9; ++mt) {
            const h16* ap = tmp + (mt * 16 + lnlo) * LDH;
            v8f c = {};
            #pragma unroll
            for (int ks = 0; ks < 4; ++ks) c = wmma16(loadA(ap, ks * 32, lane), bf[ks], c);
            int rb = mt * 16 + rsh;
            #pragma unroll
            for (int r = 0; r < 8; ++r) he[(rb + r) * LDH + col] = (h16)(c[r] + bv);
        }
    }
    __syncthreads();

    // ---- 4 message-passing layers, fully LDS-resident ----
    for (int l = 0; l < LAYERS; ++l) {
        const h16* W1 = eW1p + l * (12 * 8 * 512);
        const h16* W2 = eW2p + l * (4 * 8 * 512);
        const float* B1 = eb1 + l * HDIM;
        const float* B2 = eb2 + l * HDIM;
        const int nt = wv, col = nt * 16 + lnlo;

        // prefetch next layer's big weight blocks into L2/WGP$
        if (l + 1 < LAYERS) {
            const char* p1 = (const char*)(eW1p + (l + 1) * (12 * 8 * 512));
            const char* p2 = (const char*)(nW1p + (l + 1) * (8 * 8 * 512));
            for (int off = tid * 256; off < 12 * 8 * 512 * 2; off += 256 * 256)
                __builtin_prefetch(p1 + off, 0, 1);
            for (int off = tid * 256; off < 8 * 8 * 512 * 2; off += 256 * 256)
                __builtin_prefetch(p2 + off, 0, 1);
        }

        // edge MLP gemm1: relu([hn[src],hn[dst],he] @ W1[384,128] + b1)
        {
            v16h bf[12];
            #pragma unroll
            for (int ks = 0; ks < 12; ++ks) bf[ks] = loadB(W1 + (ks * 8 + nt) * 512, lane);
            const float bv = B1[col];
            for (int mt = 0; mt < 9; ++mt) {
                int er = mt * 16 + lnlo;
                int g = er / EPG, t = er % EPG;
                const h16* aS = hn + (g * SENS + srcN[t]) * LDH;
                const h16* aD = hn + (g * SENS + dstN[t]) * LDH;
                const h16* aE = he + er * LDH;
                v8f c = {};
                #pragma unroll
                for (int ks = 0; ks < 12; ++ks) {
                    const h16* ap = (ks < 4) ? aS : (ks < 8) ? aD : aE;
                    c = wmma16(loadA(ap, (ks & 3) * 32, lane), bf[ks], c);
                }
                int rb = mt * 16 + rsh;
                #pragma unroll
                for (int r = 0; r < 8; ++r)
                    tmp[(rb + r) * LDH + col] = (h16)fmaxf(c[r] + bv, 0.f);
            }
        }
        __syncthreads();

        // edge MLP gemm2 + residual: he += tmp @ W2[128,128] + b2
        {
            v16h bf[4];
            #pragma unroll
            for (int ks = 0; ks < 4; ++ks) bf[ks] = loadB(W2 + (ks * 8 + nt) * 512, lane);
            const float bv = B2[col];
            for (int mt = 0; mt < 9; ++mt) {
                const h16* ap = tmp + (mt * 16 + lnlo) * LDH;
                v8f c = {};
                #pragma unroll
                for (int ks = 0; ks < 4; ++ks) c = wmma16(loadA(ap, ks * 32, lane), bf[ks], c);
                int rb = mt * 16 + rsh;
                #pragma unroll
                for (int r = 0; r < 8; ++r) {
                    int o = (rb + r) * LDH + col;
                    he[o] = (h16)((float)he[o] + c[r] + bv);
                }
            }
        }
        __syncthreads();

        // aggregation: agg[n] = (graph_sum(hn) - hn[n]) / 8   (deg == 8)
        for (int idx = tid; idx < GPB * HDIM; idx += 256) {
            int g = idx >> 7, c2 = idx & 127;
            float s = 0.f;
            #pragma unroll
            for (int n = 0; n < SENS; ++n) s += (float)hn[(g * SENS + n) * LDH + c2];
            #pragma unroll
            for (int n = 0; n < SENS; ++n) {
                float v = (s - (float)hn[(g * SENS + n) * LDH + c2]) * 0.125f;
                agg[(g * SENS + n) * LDH + c2] = (h16)v;
            }
        }
        __syncthreads();

        // node MLP gemm1: relu([hn,agg] @ nW1[256,128] + b1)
        const h16* NW1 = nW1p + l * (8 * 8 * 512);
        const h16* NW2 = nW2p + l * (4 * 8 * 512);
        const float* NB1 = nb1 + l * HDIM;
        const float* NB2 = nb2 + l * HDIM;
        {
            v16h bf[8];
            #pragma unroll
            for (int ks = 0; ks < 8; ++ks) bf[ks] = loadB(NW1 + (ks * 8 + nt) * 512, lane);
            const float bv = NB1[col];
            for (int mt = 0; mt < 2; ++mt) {
                int row = mt * 16 + lnlo;
                v8f c = {};
                #pragma unroll
                for (int ks = 0; ks < 8; ++ks) {
                    const h16* ap = ((ks < 4) ? hn : agg) + row * LDH;
                    c = wmma16(loadA(ap, (ks & 3) * 32, lane), bf[ks], c);
                }
                int rb = mt * 16 + rsh;
                #pragma unroll
                for (int r = 0; r < 8; ++r)
                    tmp[(rb + r) * LDH + col] = (h16)fmaxf(c[r] + bv, 0.f);
            }
        }
        __syncthreads();

        // node MLP gemm2 + residual: hn += tmp @ nW2[128,128] + b2
        {
            v16h bf[4];
            #pragma unroll
            for (int ks = 0; ks < 4; ++ks) bf[ks] = loadB(NW2 + (ks * 8 + nt) * 512, lane);
            const float bv = NB2[col];
            for (int mt = 0; mt < 2; ++mt) {
                const h16* ap = tmp + (mt * 16 + lnlo) * LDH;
                v8f c = {};
                #pragma unroll
                for (int ks = 0; ks < 4; ++ks) c = wmma16(loadA(ap, ks * 32, lane), bf[ks], c);
                int rb = mt * 16 + rsh;
                #pragma unroll
                for (int r = 0; r < 8; ++r) {
                    int row = rb + r;
                    if (row < NB)
                        hn[row * LDH + col] = (h16)((float)hn[row * LDH + col] + c[r] + bv);
                }
            }
        }
        __syncthreads();
    }

    // ---- decoder gemm: relu(he @ dW1[128,64] + b1) (WMMA) ----
    // 4 N-tiles: wave pair (wv&3) owns nt; wv<4 does even mt, wv>=4 odd mt.
    {
        const int nt = wv & 3, col = nt * 16 + lnlo;
        v16h bf[4];
        #pragma unroll
        for (int ks = 0; ks < 4; ++ks) bf[ks] = loadB(dW1p + (ks * 4 + nt) * 512, lane);
        const float bv = db1[col];
        for (int mt = (wv >> 2); mt < 9; mt += 2) {
            const h16* ap = he + (mt * 16 + lnlo) * LDH;
            v8f c = {};
            #pragma unroll
            for (int ks = 0; ks < 4; ++ks) c = wmma16(loadA(ap, ks * 32, lane), bf[ks], c);
            int rb = mt * 16 + rsh;
            #pragma unroll
            for (int r = 0; r < 8; ++r)
                tmp[(rb + r) * LDH + col] = (h16)fmaxf(c[r] + bv, 0.f);
        }
    }
    __syncthreads();

    // ---- final [64]->1 dot + sigmoid per edge ----
    if (tid < EB) {
        float s = db2[0];
        #pragma unroll
        for (int c2 = 0; c2 < 64; ++c2) s += (float)tmp[tid * LDH + c2] * dw2[c2];
        decL[tid] = 1.f / (1.f + expf(-s));
    }
    __syncthreads();

    // ---- pair-mean over the two directions -> out[B,36] ----
    if (tid < GPB * 36) {
        int g = tid / 36, p = tid % 36;
        float v = 0.5f * (decL[g * EPG + 2 * p] + decL[g * EPG + 2 * p + 1]);
        out[(gb + g) * 36 + p] = v;
    }
}

extern "C" void kernel_launch(void* const* d_in, const int* in_sizes, int n_in,
                              void* d_out, int out_size, void* d_ws, size_t ws_size,
                              hipStream_t stream) {
    (void)in_sizes; (void)n_in; (void)out_size; (void)ws_size;
    const float* xn     = (const float*)d_in[0];   // x_nodes [N,2]
    const float* dmg    = (const float*)d_in[1];   // damage_locs [B,2]
    const float* encNw  = (const float*)d_in[2];
    const float* encNb  = (const float*)d_in[3];
    const float* encEw1 = (const float*)d_in[4];
    const float* encEb1 = (const float*)d_in[5];
    const float* encEw2 = (const float*)d_in[6];
    const float* encEb2 = (const float*)d_in[7];
    const float* ew1    = (const float*)d_in[8];
    const float* eb1    = (const float*)d_in[9];
    const float* ew2    = (const float*)d_in[10];
    const float* eb2    = (const float*)d_in[11];
    const float* nw1    = (const float*)d_in[12];
    const float* nb1    = (const float*)d_in[13];
    const float* nw2    = (const float*)d_in[14];
    const float* nb2    = (const float*)d_in[15];
    const float* dw1    = (const float*)d_in[16];
    const float* db1    = (const float*)d_in[17];
    const float* dw2    = (const float*)d_in[18];
    const float* db2    = (const float*)d_in[19];
    // d_in[20]=edge_index, d_in[21]=node_batch: topology is recomputed on-chip.

    h16* ws = (h16*)d_ws;
    // Pack all WMMA B-operands to fragment-major f16 (deterministic, re-done
    // every call; weights ~0.95 MB, stays hot in L2).
    pack_w_kernel<<<64, 256, 0, stream>>>(encEw2, ws + OFF_ENCW2, 128, 128, 1);
    pack_w_kernel<<<256, 256, 0, stream>>>(ew1, ws + OFF_EW1, 384, 128, 4);
    pack_w_kernel<<<128, 256, 0, stream>>>(ew2, ws + OFF_EW2, 128, 128, 4);
    pack_w_kernel<<<192, 256, 0, stream>>>(nw1, ws + OFF_NW1, 256, 128, 4);
    pack_w_kernel<<<128, 256, 0, stream>>>(nw2, ws + OFF_NW2, 128, 128, 4);
    pack_w_kernel<<<32, 256, 0, stream>>>(dw1, ws + OFF_DW1, 128, 64, 1);

    // 4096 graphs / 2 per block = 2048 workgroups of 8 wave32.
    gnn_fused_kernel<<<2048, 256, SMEM_BYTES, stream>>>(
        xn, dmg, encNw, encNb, encEw1, encEb1, encEb2,
        eb1, eb2, nb1, nb2, db1, dw2, db2,
        ws + OFF_ENCW2, ws + OFF_EW1, ws + OFF_EW2,
        ws + OFF_NW1, ws + OFF_NW2, ws + OFF_DW1,
        (float*)d_out);
}
